// Attention_26388279067013
// MI455X (gfx1250) — matrix operand order repaired and checked by reading, not compiled, not run on hardware
//
#include <hip/hip_runtime.h>
#include <hip/hip_bf16.h>

#define B_ 32
#define S_ 4096
#define H_ 512
#define D_ 256
#define WROW 1280   // 2H + D

typedef __attribute__((ext_vector_type(16))) __bf16 v16bf;
typedef __attribute__((ext_vector_type(8)))  float  v8f;
typedef __attribute__((ext_vector_type(4)))  unsigned int v4u;
typedef __attribute__((ext_vector_type(8)))  int v8i;
typedef __attribute__((ext_vector_type(4)))  int v4i;

#if defined(__AMDGCN__) && __has_builtin(__builtin_amdgcn_tensor_load_to_lds)
#define USE_TDM 1
#else
#define USE_TDM 0
#endif

union Frag16 {
    unsigned short us[16];
    uint4          u4[2];
    v16bf          v;
};

static __device__ __forceinline__ unsigned short f32_to_bf16(float f) {
    unsigned int u = __float_as_uint(f);
    unsigned int r = (u + 0x7FFFu + ((u >> 16) & 1u)) >> 16;  // RNE
    return (unsigned short)r;
}

// ---------------------------------------------------------------------------
// Kernel 1: convert We (W[:, :H]) to bf16; pack wv[o] = {sum_h Wc[o,h], v_w[o]}
// grid = H blocks, 256 threads
// ---------------------------------------------------------------------------
__global__ void prep_we_kernel(const float* __restrict__ W,
                               const float* __restrict__ v_w,
                               unsigned short* __restrict__ we_bf,
                               float2* __restrict__ wv) {
    const int o = blockIdx.x;
    const int t = threadIdx.x;
    const float* wr = W + (size_t)o * WROW;

    #pragma unroll
    for (int h = t; h < H_; h += 256)
        we_bf[o * H_ + h] = f32_to_bf16(wr[h]);

    float p = wr[H_ + D_ + t] + wr[H_ + D_ + t + 256];
    __shared__ float red[256];
    red[t] = p;
    __syncthreads();
    for (int off = 128; off > 0; off >>= 1) {
        if (t < off) red[t] += red[t + off];
        __syncthreads();
    }
    if (t == 0) wv[o] = make_float2(red[0], v_w[o]);
}

// ---------------------------------------------------------------------------
// Kernel 2: dec_proj[b,o] = sum_d dec[b,d] * Ws[o,d] + bias[o]
// ---------------------------------------------------------------------------
__global__ void prep_dec_kernel(const float* __restrict__ dec,
                                const float* __restrict__ W,
                                const float* __restrict__ bias,
                                float* __restrict__ dec_proj) {
    const int b = blockIdx.x;
    const int t = threadIdx.x;
    const float4* dv = (const float4*)(dec + (size_t)b * D_);

    #pragma unroll
    for (int o = t; o < H_; o += 256) {
        const float4* wr = (const float4*)(W + (size_t)o * WROW + H_);
        float s = bias[o];
        #pragma unroll 4
        for (int d = 0; d < D_ / 4; ++d) {
            float4 a = dv[d];
            float4 w = wr[d];
            s += a.x * w.x + a.y * w.y + a.z * w.z + a.w * w.w;
        }
        dec_proj[b * H_ + o] = s;
    }
}

// ---------------------------------------------------------------------------
// Kernel 3: main logits kernel. grid=(S/128,B), block=256 (8 waves).
// Wave w owns 16 s-rows; We tiles staged to LDS by the Tensor Data Mover,
// double-buffered against the 16-deep WMMA chain. B-fragments are software
// pipelined through a register double buffer to hide DS latency.
// ---------------------------------------------------------------------------
__global__ void __launch_bounds__(256, 1)
attn_logits_kernel(const float* __restrict__ enc,
                   const float* __restrict__ cov,
                   const unsigned short* __restrict__ we_bf,
                   const float* __restrict__ dec_proj,
                   const float2* __restrict__ wv,
                   const float* __restrict__ v_b,
                   float* __restrict__ logits) {
    const int b    = blockIdx.y;
    const int wave = threadIdx.x >> 5;
    const int lane = threadIdx.x & 31;
    const int half = lane >> 4;
    const int n    = lane & 15;
    const int s0   = blockIdx.x * 128 + wave * 16;

    __shared__ uint4 sB[2048];   // 2 x 16KB double buffer: 512(K) x 16(N) bf16

    // ---- A fragments (16 rows x 512 K, bf16) built once, held in VGPRs.
    const float4* rowp =
        (const float4*)(enc + ((size_t)b * S_ + (size_t)(s0 + n)) * H_);
    Frag16 a[16];
    #pragma unroll
    for (int c = 0; c < 16; ++c) {
        const int kb = c * 32 + 8 * half;
        float4 f0 = rowp[kb / 4 + 0];
        float4 f1 = rowp[kb / 4 + 1];
        float4 f2 = rowp[kb / 4 + 4];
        float4 f3 = rowp[kb / 4 + 5];
        a[c].us[0]  = f32_to_bf16(f0.x); a[c].us[1]  = f32_to_bf16(f0.y);
        a[c].us[2]  = f32_to_bf16(f0.z); a[c].us[3]  = f32_to_bf16(f0.w);
        a[c].us[4]  = f32_to_bf16(f1.x); a[c].us[5]  = f32_to_bf16(f1.y);
        a[c].us[6]  = f32_to_bf16(f1.z); a[c].us[7]  = f32_to_bf16(f1.w);
        a[c].us[8]  = f32_to_bf16(f2.x); a[c].us[9]  = f32_to_bf16(f2.y);
        a[c].us[10] = f32_to_bf16(f2.z); a[c].us[11] = f32_to_bf16(f2.w);
        a[c].us[12] = f32_to_bf16(f3.x); a[c].us[13] = f32_to_bf16(f3.y);
        a[c].us[14] = f32_to_bf16(f3.z); a[c].us[15] = f32_to_bf16(f3.w);
    }

    float covr[8];
    #pragma unroll
    for (int r = 0; r < 8; ++r)
        covr[r] = cov[(size_t)b * S_ + s0 + r + 8 * half];

    float attp[8] = {0.f, 0.f, 0.f, 0.f, 0.f, 0.f, 0.f, 0.f};

#if USE_TDM
    // LDS byte offset of the staging buffer (wave-relative, what the D# wants)
    const unsigned lds_base =
        (unsigned)(__UINTPTR_TYPE__)(__attribute__((address_space(3))) const void*)&sB[0];

    const v4i gz4 = { 0, 0, 0, 0 };
    const v8i gz8 = { 0, 0, 0, 0, 0, 0, 0, 0 };

    // Issue TDM for tile 0 into buffer 0 (one op per workgroup: wave 0).
    if (wave == 0) {
        const unsigned long long ga0 = (unsigned long long)(__UINTPTR_TYPE__)we_bf;
        v4u g0 = { 1u,                       // count=1 (valid user descriptor)
                   lds_base,                 // lds_addr
                   (unsigned)ga0,            // global_addr[31:0]
                   (unsigned)((ga0 >> 32) & 0x1FFFFFFu) | (2u << 30) }; // type=2
        v8i g1 = { 0x00010000,               // data_size=1 (2 bytes)
                   (int)(512u << 16),        // tensor_dim0 = 512 (lo16 @ [63:48])
                   (int)(512u << 16),        // tensor_dim1 = 512 (lo16 @ [95:80])
                   (int)(512u << 16),        // tile_dim0   = 512 (@ [127:112])
                   16,                       // tile_dim1   = 16  (@ [143:128])
                   512,                      // tensor_dim0_stride = 512
                   0, 0 };
        __builtin_amdgcn_tensor_load_to_lds(g0, g1, gz4, gz4, gz8, 0);
    }
#endif

    for (int ot = 0; ot < 32; ++ot) {
        const int cur = ot & 1;

#if USE_TDM
        __syncthreads();   // all waves done reading buf[cur^1] (prev-prev tile)
        if (wave == 0) {
            if (ot + 1 < 32) {
                const unsigned long long ga = (unsigned long long)(__UINTPTR_TYPE__)
                    (we_bf + (size_t)(ot + 1) * 16 * H_);
                v4u g0 = { 1u,
                           lds_base + (unsigned)((cur ^ 1) * 16384),
                           (unsigned)ga,
                           (unsigned)((ga >> 32) & 0x1FFFFFFu) | (2u << 30) };
                v8i g1 = { 0x00010000,
                           (int)(512u << 16),
                           (int)(512u << 16),
                           (int)(512u << 16),
                           16,
                           512,
                           0, 0 };
                __builtin_amdgcn_tensor_load_to_lds(g0, g1, gz4, gz4, gz8, 0);
                __builtin_amdgcn_s_wait_tensorcnt((short)1); // tile ot landed
            } else {
                __builtin_amdgcn_s_wait_tensorcnt((short)0);
            }
        }
        __syncthreads();   // buf[cur] visible to every wave
#else
        __syncthreads();
        {
            const uint4* gB = (const uint4*)(we_bf + (size_t)ot * 16 * H_);
            #pragma unroll
            for (int j = 0; j < 4; ++j)
                sB[cur * 1024 + threadIdx.x + j * 256] = gB[threadIdx.x + j * 256];
        }
        __syncthreads();
#endif

        // ---- 16-deep WMMA chain; B-fragments pipelined one chunk ahead.
        v8f acc = {};
        const int base = cur * 1024 + n * 64 + half;
        Frag16 bf_cur, bf_nxt;
        bf_cur.u4[0] = sB[base];
        bf_cur.u4[1] = sB[base + 2];
        bf_nxt = bf_cur;
        #pragma unroll
        for (int c = 0; c < 16; ++c) {
            if (c + 1 < 16) {
                bf_nxt.u4[0] = sB[base + (c + 1) * 4];
                bf_nxt.u4[1] = sB[base + (c + 1) * 4 + 2];
            }
            acc = __builtin_amdgcn_wmma_f32_16x16x32_bf16(
                false, a[c].v, false, bf_cur.v, (short)0, acc, false, false);
            bf_cur = bf_nxt;
        }

        const int    o_n = ot * 16 + n;
        const float  dp  = dec_proj[b * H_ + o_n];
        const float2 wv2 = wv[o_n];
        #pragma unroll
        for (int r = 0; r < 8; ++r) {
            float f = acc[r] + dp + covr[r] * wv2.x;
            attp[r] += wv2.y * tanhf(f);
        }
    }

    const float vb = v_b[0];
    #pragma unroll
    for (int r = 0; r < 8; ++r) {
        float v = attp[r];
        v += __shfl_xor(v, 1, 32);
        v += __shfl_xor(v, 2, 32);
        v += __shfl_xor(v, 4, 32);
        v += __shfl_xor(v, 8, 32);
        if (n == 0)
            logits[(size_t)b * S_ + s0 + r + 8 * half] = v + vb;
    }
}

// ---------------------------------------------------------------------------
// Kernel 4: masked softmax over S + new coverage. grid = B, block = 256.
// ---------------------------------------------------------------------------
__global__ void softmax_kernel(const float* __restrict__ logits,
                               const float* __restrict__ cov,
                               const int* __restrict__ lens,
                               float* __restrict__ out_w,
                               float* __restrict__ out_c) {
    const int b = blockIdx.x;
    const int t = threadIdx.x;
    const int len = lens[b];

    float xv[16];
    float m = -INFINITY;
    #pragma unroll
    for (int j = 0; j < 16; ++j) {
        const int s = t + j * 256;
        float x = (s < len) ? logits[(size_t)b * S_ + s] : -INFINITY;
        xv[j] = x;
        m = fmaxf(m, x);
    }

    __shared__ float red[256];
    red[t] = m;
    __syncthreads();
    for (int off = 128; off > 0; off >>= 1) {
        if (t < off) red[t] = fmaxf(red[t], red[t + off]);
        __syncthreads();
    }
    m = red[0];
    __syncthreads();

    float sum = 0.f;
    #pragma unroll
    for (int j = 0; j < 16; ++j) {
        float e = (xv[j] == -INFINITY) ? 0.f : __expf(xv[j] - m);
        xv[j] = e;
        sum += e;
    }
    red[t] = sum;
    __syncthreads();
    for (int off = 128; off > 0; off >>= 1) {
        if (t < off) red[t] += red[t + off];
        __syncthreads();
    }
    const float inv = 1.0f / red[0];

    #pragma unroll
    for (int j = 0; j < 16; ++j) {
        const int s = t + j * 256;
        const float w = xv[j] * inv;
        out_w[(size_t)b * S_ + s] = w;
        out_c[(size_t)b * S_ + s] = cov[(size_t)b * S_ + s] + w;
    }
}

// ---------------------------------------------------------------------------
extern "C" void kernel_launch(void* const* d_in, const int* in_sizes, int n_in,
                              void* d_out, int out_size, void* d_ws, size_t ws_size,
                              hipStream_t stream) {
    const float* dec  = (const float*)d_in[0];
    const float* enc  = (const float*)d_in[1];
    const float* cov  = (const float*)d_in[2];
    const int*   lens = (const int*)d_in[3];
    const float* W    = (const float*)d_in[4];
    const float* bias = (const float*)d_in[5];
    const float* v_w  = (const float*)d_in[6];
    const float* v_b  = (const float*)d_in[7];

    float* out_w = (float*)d_out;
    float* out_c = out_w + (size_t)B_ * S_;

    char* ws = (char*)d_ws;
    unsigned short* we_bf    = (unsigned short*)ws;                  // 512 KB
    float*          dec_proj = (float*)(ws + 524288);                // 64 KB
    float2*         wv       = (float2*)(ws + 524288 + 65536);       // 4 KB
    float*          logits   = (float*)(ws + 524288 + 65536 + 4096); // 512 KB

    prep_we_kernel<<<H_, 256, 0, stream>>>(W, v_w, we_bf, wv);
    prep_dec_kernel<<<B_, 256, 0, stream>>>(dec, W, bias, dec_proj);

    dim3 grid(S_ / 128, B_);
    attn_logits_kernel<<<grid, 256, 0, stream>>>(enc, cov, we_bf, dec_proj,
                                                 wv, v_b, logits);

    softmax_kernel<<<B_, 256, 0, stream>>>(logits, cov, lens, out_w, out_c);
}
